// MultiHeadSelfAttentionModule_6193342840934
// MI455X (gfx1250) — matrix operand order repaired and checked by reading, not compiled, not run on hardware
//
#include <hip/hip_runtime.h>

// ---------------------------------------------------------------------------
// Fused pre-LN multi-head self-attention for MI455X (gfx1250, wave32, WMMA).
// B=2048, S=32, D=1024, H=16, DK=64.
// Memory-bound (~0.55 GB min traffic vs 0.56 TFLOP): single fused kernel,
// one HBM pass over x, weights in f16 (L2-resident), intermediates in LDS.
// This revision: keep weight-pointer provenance (global_load_b128, not flat),
// hoist per-lane weight base so K-loop loads use immediate offsets.
// ---------------------------------------------------------------------------

typedef __attribute__((ext_vector_type(16))) _Float16 v16h;
typedef __attribute__((ext_vector_type(8)))  _Float16 v8h;
typedef __attribute__((ext_vector_type(4)))  _Float16 v4h;
typedef __attribute__((ext_vector_type(8)))  float    v8f;
typedef __attribute__((ext_vector_type(4)))  float    v4f;

#define DI __device__ __forceinline__

constexpr int B_  = 2048;
constexpr int S_  = 32;
constexpr int D_  = 1024;
constexpr int H_  = 16;
constexpr int DK_ = 64;

// LDS layout (units: _Float16 elements). Strides padded to spread banks.
constexpr int XS  = 1032;                 // row stride for 32 x 1024 buffers
constexpr int KTS = 40;                   // K^T row stride (32 s-values + pad)
constexpr int ATS = 40;                   // attn row stride (32 cols + pad)
constexpr int XN_OFF = 0;                 // xn  [32][XS]  (reused as ctx)
constexpr int Q_OFF  = XN_OFF + 32 * XS;  // q   [32][XS]
constexpr int V_OFF  = Q_OFF  + 32 * XS;  // v   [32][XS]
constexpr int KT_OFF = V_OFF  + 32 * XS;  // kT  [16][64][KTS]
constexpr int AT_OFF = KT_OFF + H_ * DK_ * KTS; // attn [16][32][ATS]
constexpr int SMEM_HALFS = AT_OFF + H_ * S_ * ATS;
constexpr size_t SMEM_BYTES = (size_t)SMEM_HALFS * sizeof(_Float16); // ~313 KB

// ---- WMMA fragment helpers (layouts per CDNA5 ISA §7.12.2) -----

// A-matrix 16x32 f16: lane L -> row m=L&15; VGPR0-3 hold K = hi*8..hi*8+7,
// VGPR4-7 hold K = 16+hi*8..16+hi*8+7 where hi = L>>4. Two 16B loads.
DI v16h load_A(const _Float16* tile, int stride, int lane) {
  const int m  = lane & 15;
  const int hi = lane >> 4;
  const _Float16* p = tile + m * stride + hi * 8;
  union { v16h v; v8h h[2]; } u;
  u.h[0] = *(const v8h*)(p);
  u.h[1] = *(const v8h*)(p + 16);
  return u.v;
}

// B-matrix 32x16 f16: lane L holds column N=L&15; K order matches load_A.
DI v16h load_B(const _Float16* tile, int stride, int lane) {
  const _Float16* p = tile + (lane & 15) + (lane >> 4) * 8 * stride;
  v16h v;
#pragma unroll
  for (int i = 0; i < 8; ++i) {
    v[i]     = p[i * stride];
    v[8 + i] = p[(16 + i) * stride];
  }
  return v;
}

DI v8f wmma_f16(v16h a, v16h b, v8f c) {
  return __builtin_amdgcn_wmma_f32_16x16x32_f16(false, a, false, b,
                                                (short)0, c, false, false);
}

// ---------------------------------------------------------------------------
// Kernel 0: convert the four 1024x1024 fp32 weight matrices to f16 in d_ws.
// 8 MB total -> stays resident in the 192 MB L2 for the whole main grid.
// ---------------------------------------------------------------------------
__global__ void cvt_w_kernel(const float* __restrict__ wq,
                             const float* __restrict__ wk,
                             const float* __restrict__ wv,
                             const float* __restrict__ wo,
                             _Float16* __restrict__ w16) {
  const int n = D_ * D_;
  int i = blockIdx.x * blockDim.x + threadIdx.x;
  const int st = gridDim.x * blockDim.x;
  for (; i < n; i += st) {
    w16[i        ] = (_Float16)wq[i];
    w16[i + n    ] = (_Float16)wk[i];
    w16[i + 2 * n] = (_Float16)wv[i];
    w16[i + 3 * n] = (_Float16)wo[i];
  }
}

// ---------------------------------------------------------------------------
// Q/K/V projection GEMM: out(32x128 per wave) = xn(32x1024) @ W(1024x128)+b.
// Templated on destination so each call site keeps a *direct* pointer chain
// from the kernel argument (=> global_load_b128 with immediate offsets).
// MODE: 0 = Q (row-major), 1 = K (store transposed per head), 2 = V.
// ---------------------------------------------------------------------------
template <int MODE>
DI void qkv_gemm(const _Float16* __restrict__ W, const float* __restrict__ bs,
                 const _Float16* xn, _Float16* qs, _Float16* kts, _Float16* vs,
                 int wave, int lane) {
  const int nb = wave * 128;
#pragma unroll 1
  for (int nt = 0; nt < 8; ++nt) {
    const int n0 = nb + nt * 16;
    // B fragment gathered per column; k-step advances by 32*D_ halfs = 64 KB,
    // which fits the signed 24-bit global IOFFSET as an immediate.
    const _Float16* Wl = W + n0;
    v8f c0 = {}, c1 = {};
#pragma unroll 4
    for (int k = 0; k < 32; ++k) {
      const int k0 = k * 32;
      v16h a0  = load_A(xn + k0, XS, lane);
      v16h a1  = load_A(xn + 16 * XS + k0, XS, lane);
      v16h bfr = load_B(Wl + (size_t)k0 * D_, D_, lane);
      c0 = wmma_f16(a0, bfr, c0);
      c1 = wmma_f16(a1, bfr, c1);
    }
    const int   nc    = n0 + (lane & 15);
    const float bvv   = bs[nc];
    const int   rbase = (lane >> 4) ? 8 : 0;
#pragma unroll
    for (int r = 0; r < 8; ++r) {
      const int r0 = rbase + r;
      const _Float16 h0 = (_Float16)(c0[r] + bvv);
      const _Float16 h1 = (_Float16)(c1[r] + bvv);
      if (MODE == 0) {
        qs[r0 * XS + nc]        = h0;
        qs[(16 + r0) * XS + nc] = h1;
      } else if (MODE == 1) {
        const int hd = nc >> 6, kd = nc & 63;
        kts[hd * (DK_ * KTS) + kd * KTS + r0]      = h0;
        kts[hd * (DK_ * KTS) + kd * KTS + 16 + r0] = h1;
      } else {
        vs[r0 * XS + nc]        = h0;
        vs[(16 + r0) * XS + nc] = h1;
      }
    }
  }
}

// ---------------------------------------------------------------------------
// Main fused kernel: one workgroup per batch element (grid = 2048).
// 256 threads = 8 wave32. Wave w owns output-column block [w*128, w*128+128).
// ---------------------------------------------------------------------------
__global__ void __launch_bounds__(256)
mhsa_kernel(const float* __restrict__ x,
            const float* __restrict__ ln_g,
            const float* __restrict__ ln_b,
            const float* __restrict__ bq,
            const float* __restrict__ bk,
            const float* __restrict__ bv,
            const float* __restrict__ bo,
            const float* __restrict__ rel_bias,
            const _Float16* __restrict__ w16,
            float* __restrict__ out) {
  extern __shared__ _Float16 smem[];
  _Float16* xn  = smem + XN_OFF;   // layernormed x, f16 (later: ctx)
  _Float16* qs  = smem + Q_OFF;    // Q, row-major [s][d]
  _Float16* vs  = smem + V_OFF;    // V, row-major [s][d]
  _Float16* kts = smem + KT_OFF;   // K^T per head [h][dk][s]
  _Float16* ats = smem + AT_OFF;   // attn per head [h][s_q][s_k]

  const int tid  = threadIdx.x;
  const int wave = tid >> 5;
  const int lane = tid & 31;
  const int b    = blockIdx.x;
  const float* xb = x + (size_t)b * (S_ * D_);

  // ---- Phase A: LayerNorm -> xn (f16 in LDS). One wave per 4 rows. ----
  for (int rr = 0; rr < 4; ++rr) {
    const int row = wave * 4 + rr;
    const float* xr = xb + row * D_;
    v4f vals[8];
    float s = 0.f, s2 = 0.f;
#pragma unroll
    for (int c = 0; c < 8; ++c) {
      vals[c] = *(const v4f*)(xr + c * 128 + lane * 4);
#pragma unroll
      for (int j = 0; j < 4; ++j) { float t = vals[c][j]; s += t; s2 += t * t; }
    }
#pragma unroll
    for (int off = 16; off > 0; off >>= 1) {
      s  += __shfl_xor(s,  off, 32);
      s2 += __shfl_xor(s2, off, 32);
    }
    const float mean = s * (1.f / D_);
    const float var  = s2 * (1.f / D_) - mean * mean;
    const float rstd = rsqrtf(var + 1e-5f);
#pragma unroll
    for (int c = 0; c < 8; ++c) {
      const int col = c * 128 + lane * 4;
      v4f g  = *(const v4f*)(ln_g + col);
      v4f be = *(const v4f*)(ln_b + col);
      v4h h;
#pragma unroll
      for (int j = 0; j < 4; ++j)
        h[j] = (_Float16)((vals[c][j] - mean) * rstd * g[j] + be[j]);
      *(v4h*)(xn + row * XS + col) = h;
    }
  }
  __syncthreads();

  // ---- Phase B: Q/K/V projections (32x1024x1024 each, WMMA f16) ----
  qkv_gemm<0>(w16,               bq, xn, qs, kts, vs, wave, lane);
  qkv_gemm<1>(w16 +     D_ * D_, bk, xn, qs, kts, vs, wave, lane);
  qkv_gemm<2>(w16 + 2 * D_ * D_, bv, xn, qs, kts, vs, wave, lane);
  __syncthreads();

  // ---- Phase C: scores = Q K^T / 8 + rel_bias; softmax -> attn (f16) ----
  // 2 heads per wave; scores live in registers, softmax via lane shuffles.
  for (int hh = 0; hh < 2; ++hh) {
    const int h = wave * 2 + hh;
    v8f cc00 = {}, cc01 = {}, cc10 = {}, cc11 = {};
#pragma unroll
    for (int ks = 0; ks < 2; ++ks) {
      const int k0 = ks * 32;
      v16h a0 = load_A(qs + h * DK_ + k0, XS, lane);
      v16h a1 = load_A(qs + 16 * XS + h * DK_ + k0, XS, lane);
      v16h b0 = load_B(kts + h * (DK_ * KTS) + k0 * KTS + 0, KTS, lane);
      v16h b1 = load_B(kts + h * (DK_ * KTS) + k0 * KTS + 16, KTS, lane);
      cc00 = wmma_f16(a0, b0, cc00);
      cc01 = wmma_f16(a0, b1, cc01);
      cc10 = wmma_f16(a1, b0, cc10);
      cc11 = wmma_f16(a1, b1, cc11);
    }
    const float* rb = rel_bias + h * (S_ * S_);
    const int cl    = lane & 15;
    const int rsel  = (lane >> 4) ? 8 : 0;
#pragma unroll
    for (int mt = 0; mt < 2; ++mt) {
      const v8f& cA = mt ? cc10 : cc00;  // cols 0..15
      const v8f& cB = mt ? cc11 : cc01;  // cols 16..31
#pragma unroll
      for (int r = 0; r < 8; ++r) {
        const int row = mt * 16 + rsel + r;
        float v0 = cA[r] * 0.125f + rb[row * 32 + cl];
        float v1 = cB[r] * 0.125f + rb[row * 32 + 16 + cl];
        float mx = fmaxf(v0, v1);
#pragma unroll
        for (int off = 8; off > 0; off >>= 1)
          mx = fmaxf(mx, __shfl_xor(mx, off, 32));
        const float e0 = __expf(v0 - mx);
        const float e1 = __expf(v1 - mx);
        float sm = e0 + e1;
#pragma unroll
        for (int off = 8; off > 0; off >>= 1)
          sm += __shfl_xor(sm, off, 32);
        const float inv = 1.f / sm;
        ats[h * (S_ * ATS) + row * ATS + cl]      = (_Float16)(e0 * inv);
        ats[h * (S_ * ATS) + row * ATS + 16 + cl] = (_Float16)(e1 * inv);
      }
    }
  }
  __syncthreads();

  // ---- Phase D: ctx = attn @ V (K=32 -> single WMMA step). ctx reuses xn. --
  _Float16* ctx = xn;
  {
    const int nb = wave * 128;
#pragma unroll 1
    for (int nt = 0; nt < 8; ++nt) {
      const int n0 = nb + nt * 16;
      const int h  = n0 >> 6;
      v16h a0  = load_A(ats + h * (S_ * ATS), ATS, lane);
      v16h a1  = load_A(ats + h * (S_ * ATS) + 16 * ATS, ATS, lane);
      v16h bfr = load_B(vs + n0, XS, lane);
      v8f c0 = {}, c1 = {};
      c0 = wmma_f16(a0, bfr, c0);
      c1 = wmma_f16(a1, bfr, c1);
      const int nc = n0 + (lane & 15);
      const int rbase = (lane >> 4) ? 8 : 0;
#pragma unroll
      for (int r = 0; r < 8; ++r) {
        ctx[(rbase + r) * XS + nc]      = (_Float16)c0[r];
        ctx[(16 + rbase + r) * XS + nc] = (_Float16)c1[r];
      }
    }
  }
  __syncthreads();

  // ---- Phase E: out = ctx @ wo + bo + x (residual), f32 to HBM ----
  {
    const int nb = wave * 128;
    float* outb = out + (size_t)b * (S_ * D_);
#pragma unroll 1
    for (int nt = 0; nt < 8; ++nt) {
      const int n0 = nb + nt * 16;
      // Direct chain from kernel arg w16 -> global_load; hoisted lane base.
      const _Float16* Wl = w16 + 3 * (size_t)D_ * D_ + n0;
      v8f c0 = {}, c1 = {};
#pragma unroll 4
      for (int k = 0; k < 32; ++k) {
        const int k0 = k * 32;
        v16h a0  = load_A(ctx + k0, XS, lane);
        v16h a1  = load_A(ctx + 16 * XS + k0, XS, lane);
        v16h bfr = load_B(Wl + (size_t)k0 * D_, D_, lane);
        c0 = wmma_f16(a0, bfr, c0);
        c1 = wmma_f16(a1, bfr, c1);
      }
      const int   nc  = n0 + (lane & 15);
      const float bov = bo[nc];
      const int   rbase = (lane >> 4) ? 8 : 0;
#pragma unroll
      for (int r = 0; r < 8; ++r) {
        const int r0 = rbase + r;
        const int r1 = 16 + rbase + r;
        outb[r0 * D_ + nc] = c0[r] + bov + xb[r0 * D_ + nc];
        outb[r1 * D_ + nc] = c1[r] + bov + xb[r1 * D_ + nc];
      }
    }
  }
}

// ---------------------------------------------------------------------------
extern "C" void kernel_launch(void* const* d_in, const int* in_sizes, int n_in,
                              void* d_out, int out_size, void* d_ws, size_t ws_size,
                              hipStream_t stream) {
  const float* x    = (const float*)d_in[0];
  const float* ln_g = (const float*)d_in[1];
  const float* ln_b = (const float*)d_in[2];
  const float* wq   = (const float*)d_in[3];
  const float* bq   = (const float*)d_in[4];
  const float* wk   = (const float*)d_in[5];
  const float* bk   = (const float*)d_in[6];
  const float* wv   = (const float*)d_in[7];
  const float* bv   = (const float*)d_in[8];
  const float* wo   = (const float*)d_in[9];
  const float* bo   = (const float*)d_in[10];
  const float* rel  = (const float*)d_in[11];
  float* out = (float*)d_out;
  _Float16* w16 = (_Float16*)d_ws;   // 4 * 1024*1024 f16 = 8 MB

  // Opt into >64KB dynamic LDS (idempotent; host-side, not a stream op).
  (void)hipFuncSetAttribute(reinterpret_cast<const void*>(&mhsa_kernel),
                            hipFuncAttributeMaxDynamicSharedMemorySize,
                            (int)SMEM_BYTES);

  cvt_w_kernel<<<dim3(2048), dim3(256), 0, stream>>>(wq, wk, wv, wo, w16);
  mhsa_kernel<<<dim3(B_), dim3(256), SMEM_BYTES, stream>>>(
      x, ln_g, ln_b, bq, bk, bv, bo, rel, w16, out);
}
